// LGCNModelBase_86457691668737
// MI455X (gfx1250) — matrix operand order repaired and checked by reading, not compiled, not run on hardware
//
#include <hip/hip_runtime.h>
#include <hip/hip_bf16.h>

// ---------------------------------------------------------------------------
// Fused LightGCN-recsys block for MI455X (gfx1250, wave32, WMMA).
//   out[:, :256] = LN( concat(e_int,e_test,e_item,e_tag, gW@graph_row+gb) @ comb_W + cb )
//   out[:, 256:] = LN( cont @ cont_W + b )
// GEMMs run as v_wmma_f32_16x16x32_bf16 (f32 accumulate), intermediates in LDS.
// Weights are pre-packed to bf16 WMMA-B-fragment layout in d_ws when it fits.
// ---------------------------------------------------------------------------

typedef __bf16 bf16_t;
typedef __attribute__((ext_vector_type(16))) __bf16 v16bf;
typedef __attribute__((ext_vector_type(8)))  __bf16 v8bf;
typedef __attribute__((ext_vector_type(4)))  __bf16 v4bf;
typedef __attribute__((ext_vector_type(8)))  float  v8f;

#define BS_TOTAL   131072     // B*S
#define INTD       256
#define HD         512
#define MT         32         // tokens per block
#define EMB_STRIDE 1288       // 1280 + 8 pad (bf16); 644 dwords -> rows hit distinct banks
#define GR_STRIDE  520        // 512 + 8 pad (bf16); 260 dwords
#define CATE_STRIDE 260       // f32 stride for LN staging
#define N_USER_OFF 7441       // N_USER - 1
#define KT_G       16         // K/32 for graph GEMM (K=512)
#define KT_C       40         // K/32 for comb GEMM  (K=1280)

__device__ __forceinline__ v8f do_wmma(v16bf a, v16bf b, v8f c) {
    return __builtin_amdgcn_wmma_f32_16x16x32_bf16(
        /*neg_a=*/false, a, /*neg_b=*/false, b,
        /*c_mod=*/(short)0, c, /*reuse_a=*/false, /*reuse_b=*/false);
}

// A fragment (16x32 bf16, M=row per lane): elems 0..7 = K kbase..kbase+7,
// elems 8..15 = K kbase+16..kbase+23  (kbase = kk*32 + h*8), per ISA 7.12.2.
__device__ __forceinline__ v16bf load_a_frag(const bf16_t* p) {
    v8bf lo = *(const v8bf*)(p);
    v8bf hi = *(const v8bf*)(p + 16);
    return __builtin_shufflevector(lo, hi, 0,1,2,3,4,5,6,7,8,9,10,11,12,13,14,15);
}

// Direct (unpacked) B fragment: lane(h,n) elem e = W[(kk*32+h*8+e+(e>=8?8:0))][nt*16+n].
__device__ __forceinline__ v16bf load_b_frag(const float* __restrict__ W,
                                             int krow, int col) {
    v16bf b;
#pragma unroll
    for (int e = 0; e < 16; ++e)
        b[e] = (bf16_t)W[(size_t)(krow - ((krow >> 4) & 1) * 8 + e + (e >= 8 ? 8 : 0)) * 256 + col];
    return b;
}

// Packed B fragment: 16 contiguous bf16 per (nt, kk, lane) -> two 16B loads.
__device__ __forceinline__ v16bf load_b_packed(const bf16_t* __restrict__ P,
                                               int Kt, int nt, int kk, int lane) {
    const bf16_t* p = P + ((size_t)((nt * Kt + kk) * 32 + lane)) * 16;
    v8bf lo = *(const v8bf*)(p);
    v8bf hi = *(const v8bf*)(p + 8);
    return __builtin_shufflevector(lo, hi, 0,1,2,3,4,5,6,7,8,9,10,11,12,13,14,15);
}

__device__ __forceinline__ void stage256(bf16_t* dst, const float* __restrict__ src,
                                         int lane) {
    const float4* s4 = (const float4*)src;
#pragma unroll
    for (int i = 0; i < 2; ++i) {
        float4 v = s4[lane + 32 * i];
        v4bf pv; pv.x = (bf16_t)v.x; pv.y = (bf16_t)v.y;
        pv.z = (bf16_t)v.z; pv.w = (bf16_t)v.w;
        *(v4bf*)(dst + (lane + 32 * i) * 4) = pv;
    }
}

// --------- weight pre-pack: W[K][256] f32 -> P[nt][kk][lane][16] bf16 --------
__global__ __launch_bounds__(256)
void pack_weights_kernel(const float* __restrict__ W, bf16_t* __restrict__ P,
                         int Kt /* K/32 */) {
    const int idx = blockIdx.x * blockDim.x + threadIdx.x;  // (nt*Kt + kk)*32 + lane
    const int total = 16 * Kt * 32;
    if (idx >= total) return;
    const int lane = idx & 31;
    const int kk   = (idx >> 5) % Kt;
    const int nt   = idx / (Kt * 32);
    const int h = lane >> 4, n = lane & 15;
    const int col  = nt * 16 + n;
    const int krow = kk * 32 + h * 8;
    bf16_t* dst = P + (size_t)idx * 16;
#pragma unroll
    for (int e = 0; e < 16; ++e)
        dst[e] = (bf16_t)W[(size_t)(krow + e + (e >= 8 ? 8 : 0)) * 256 + col];
}

template <bool PACKED>
__global__ __launch_bounds__(256)
void lgcn_fused_kernel(
    const int*   __restrict__ cate_testId,
    const int*   __restrict__ cate_itemId,
    const int*   __restrict__ cate_tag,
    const float* __restrict__ cont_feats,
    const int*   __restrict__ interaction,
    const float* __restrict__ emb_interaction,
    const float* __restrict__ emb_testId,
    const float* __restrict__ emb_itemId,
    const float* __restrict__ emb_tag,
    const float* __restrict__ graph_table,
    const float* __restrict__ graph_W,
    const float* __restrict__ graph_b,
    const float* __restrict__ comb_W,
    const float* __restrict__ comb_b,
    const float* __restrict__ comb_g,
    const float* __restrict__ comb_beta,
    const float* __restrict__ cont_W,
    const float* __restrict__ cont_b,
    const float* __restrict__ cont_g,
    const float* __restrict__ cont_beta,
    const bf16_t* __restrict__ Pg,      // packed graph_W (when PACKED)
    const bf16_t* __restrict__ Pc,      // packed comb_W  (when PACKED)
    float* __restrict__ out)
{
    extern __shared__ char smem[];
    bf16_t* sEmb = (bf16_t*)smem;                               // MT x EMB_STRIDE bf16
    float*  sCate = (float*)(smem + MT * EMB_STRIDE * 2);       // MT x 260 f32
    bf16_t* sGraph = (bf16_t*)sCate;                            // alias: MT x 520 bf16

    const int tid  = threadIdx.x;
    const int w    = tid >> 5;          // wave 0..7
    const int lane = tid & 31;
    const int h    = lane >> 4;         // lane half
    const int ln16 = lane & 15;         // M (for A/D) or N (for B/D)
    const size_t tok0 = (size_t)blockIdx.x * MT;

    // ---------------- Phase 0: gather + f32->bf16 stage into LDS ------------
#pragma unroll
    for (int rr = 0; rr < 4; ++rr) {
        const int row = w * 4 + rr;
        const size_t tok = tok0 + row;
        const int tId = cate_testId[tok];
        const int iId = cate_itemId[tok];
        const int tg  = cate_tag[tok];
        const int itn = interaction[tok];
        bf16_t* erow = sEmb + row * EMB_STRIDE;
        stage256(erow + 0 * INTD, emb_interaction + (size_t)itn * INTD, lane);
        stage256(erow + 1 * INTD, emb_testId      + (size_t)tId * INTD, lane);
        stage256(erow + 2 * INTD, emb_itemId      + (size_t)iId * INTD, lane);
        stage256(erow + 3 * INTD, emb_tag         + (size_t)tg  * INTD, lane);
        const float4* g4 = (const float4*)(graph_table + (size_t)(N_USER_OFF + tId) * HD);
        bf16_t* grow = sGraph + row * GR_STRIDE;
#pragma unroll
        for (int i = 0; i < 4; ++i) {
            float4 v = g4[lane + 32 * i];
            v4bf pv; pv.x = (bf16_t)v.x; pv.y = (bf16_t)v.y;
            pv.z = (bf16_t)v.z; pv.w = (bf16_t)v.w;
            *(v4bf*)(grow + (lane + 32 * i) * 4) = pv;
        }
    }
    __syncthreads();

    // ---------------- Phase 1: g = graph_row @ graph_W + gb  (K = 512) ------
    // Wave w owns n-tiles {2w, 2w+1}, both m-tiles; B fragments reused across m.
    {
        v8f acc[2][2];
#pragma unroll
        for (int r = 0; r < 2; ++r)
#pragma unroll
            for (int t = 0; t < 2; ++t) acc[r][t] = (v8f)0.0f;

        for (int kk = 0; kk < KT_G; ++kk) {
            v16bf af[2];
#pragma unroll
            for (int r = 0; r < 2; ++r)
                af[r] = load_a_frag(sGraph + (r * 16 + ln16) * GR_STRIDE + kk * 32 + h * 8);
#pragma unroll
            for (int t = 0; t < 2; ++t) {
                const int nt = 2 * w + t;
                v16bf bf;
                if constexpr (PACKED) bf = load_b_packed(Pg, KT_G, nt, kk, lane);
                else                  bf = load_b_frag(graph_W, kk * 32 + h * 16, nt * 16 + ln16);
#pragma unroll
                for (int r = 0; r < 2; ++r)
                    acc[r][t] = do_wmma(af[r], bf, acc[r][t]);
            }
        }
        // bias + write g (bf16) into concat-embed columns [1024, 1280)
#pragma unroll
        for (int t = 0; t < 2; ++t) {
            const int nt = 2 * w + t;
            const float gb = graph_b[nt * 16 + ln16];
#pragma unroll
            for (int r = 0; r < 2; ++r)
#pragma unroll
                for (int e = 0; e < 8; ++e)
                    sEmb[(r * 16 + e + 8 * h) * EMB_STRIDE + 1024 + nt * 16 + ln16] =
                        (bf16_t)(acc[r][t][e] + gb);
        }
    }

    // ---------------- Phase 2: cate_pre = embed @ comb_W + cb (K = 1280) ----
    // k-steps 0..31 touch only the gathered-embedding half of K (ready since the
    // first barrier), so they overlap the g-publication barrier; 32..39 wait.
    {
        v8f acc[2][2];
#pragma unroll
        for (int r = 0; r < 2; ++r)
#pragma unroll
            for (int t = 0; t < 2; ++t) acc[r][t] = (v8f)0.0f;

        for (int kk = 0; kk < 32; ++kk) {
            v16bf af[2];
#pragma unroll
            for (int r = 0; r < 2; ++r)
                af[r] = load_a_frag(sEmb + (r * 16 + ln16) * EMB_STRIDE + kk * 32 + h * 8);
#pragma unroll
            for (int t = 0; t < 2; ++t) {
                const int nt = 2 * w + t;
                v16bf bf;
                if constexpr (PACKED) bf = load_b_packed(Pc, KT_C, nt, kk, lane);
                else                  bf = load_b_frag(comb_W, kk * 32 + h * 16, nt * 16 + ln16);
#pragma unroll
                for (int r = 0; r < 2; ++r)
                    acc[r][t] = do_wmma(af[r], bf, acc[r][t]);
            }
        }
        __syncthreads();   // g columns [1024,1280) now visible from all waves
        for (int kk = 32; kk < KT_C; ++kk) {
            v16bf af[2];
#pragma unroll
            for (int r = 0; r < 2; ++r)
                af[r] = load_a_frag(sEmb + (r * 16 + ln16) * EMB_STRIDE + kk * 32 + h * 8);
#pragma unroll
            for (int t = 0; t < 2; ++t) {
                const int nt = 2 * w + t;
                v16bf bf;
                if constexpr (PACKED) bf = load_b_packed(Pc, KT_C, nt, kk, lane);
                else                  bf = load_b_frag(comb_W, kk * 32 + h * 16, nt * 16 + ln16);
#pragma unroll
                for (int r = 0; r < 2; ++r)
                    acc[r][t] = do_wmma(af[r], bf, acc[r][t]);
            }
        }
#pragma unroll
        for (int t = 0; t < 2; ++t) {
            const int nt = 2 * w + t;
            const float cb = comb_b[nt * 16 + ln16];
#pragma unroll
            for (int r = 0; r < 2; ++r)
#pragma unroll
                for (int e = 0; e < 8; ++e)
                    sCate[(r * 16 + e + 8 * h) * CATE_STRIDE + nt * 16 + ln16] =
                        acc[r][t][e] + cb;
        }
    }
    __syncthreads();

    // ---------------- Phase 3: LayerNorms + cont path + store ---------------
#pragma unroll
    for (int rr = 0; rr < 4; ++rr) {
        const int row = w * 4 + rr;
        const size_t tok = tok0 + row;

        // cate LN over 256 cols (8 per lane, wave32 xor-reduce)
        float x[8], s = 0.f, s2 = 0.f;
#pragma unroll
        for (int i = 0; i < 8; ++i) {
            x[i] = sCate[row * CATE_STRIDE + lane + 32 * i];
            s += x[i]; s2 += x[i] * x[i];
        }
#pragma unroll
        for (int m = 16; m >= 1; m >>= 1) {
            s  += __shfl_xor(s,  m, 32);
            s2 += __shfl_xor(s2, m, 32);
        }
        float mean = s * (1.f / 256.f);
        float var  = s2 * (1.f / 256.f) - mean * mean;
        float rs   = rsqrtf(var + 1e-5f);
#pragma unroll
        for (int i = 0; i < 8; ++i) {
            const int col = lane + 32 * i;
            out[tok * HD + col] = (x[i] - mean) * rs * comb_g[col] + comb_beta[col];
        }

        // cont path: [3] @ [3,256] + LN
        const float c0 = cont_feats[tok * 3 + 0];
        const float c1 = cont_feats[tok * 3 + 1];
        const float c2 = cont_feats[tok * 3 + 2];
        float y[8]; s = 0.f; s2 = 0.f;
#pragma unroll
        for (int i = 0; i < 8; ++i) {
            const int col = lane + 32 * i;
            y[i] = c0 * cont_W[col] + c1 * cont_W[256 + col] + c2 * cont_W[512 + col]
                 + cont_b[col];
            s += y[i]; s2 += y[i] * y[i];
        }
#pragma unroll
        for (int m = 16; m >= 1; m >>= 1) {
            s  += __shfl_xor(s,  m, 32);
            s2 += __shfl_xor(s2, m, 32);
        }
        mean = s * (1.f / 256.f);
        var  = s2 * (1.f / 256.f) - mean * mean;
        rs   = rsqrtf(var + 1e-5f);
#pragma unroll
        for (int i = 0; i < 8; ++i) {
            const int col = lane + 32 * i;
            out[tok * HD + 256 + col] = (y[i] - mean) * rs * cont_g[col] + cont_beta[col];
        }
    }
}

extern "C" void kernel_launch(void* const* d_in, const int* in_sizes, int n_in,
                              void* d_out, int out_size, void* d_ws, size_t ws_size,
                              hipStream_t stream) {
    (void)in_sizes; (void)n_in; (void)out_size;

    const int*   cate_testId     = (const int*)  d_in[0];
    const int*   cate_itemId     = (const int*)  d_in[1];
    const int*   cate_tag        = (const int*)  d_in[2];
    const float* cont_feats      = (const float*)d_in[3];
    const int*   interaction     = (const int*)  d_in[4];
    const float* emb_interaction = (const float*)d_in[5];
    const float* emb_testId      = (const float*)d_in[6];
    const float* emb_itemId      = (const float*)d_in[7];
    const float* emb_tag         = (const float*)d_in[8];
    const float* graph_table     = (const float*)d_in[9];
    const float* graph_W         = (const float*)d_in[10];
    const float* graph_b         = (const float*)d_in[11];
    const float* comb_W          = (const float*)d_in[12];
    const float* comb_b          = (const float*)d_in[13];
    const float* comb_g          = (const float*)d_in[14];
    const float* comb_beta       = (const float*)d_in[15];
    const float* cont_W          = (const float*)d_in[16];
    const float* cont_b          = (const float*)d_in[17];
    const float* cont_g          = (const float*)d_in[18];
    const float* cont_beta       = (const float*)d_in[19];
    float* out = (float*)d_out;

    const size_t lds_bytes = (size_t)MT * EMB_STRIDE * sizeof(bf16_t)   // 82,432
                           + (size_t)MT * CATE_STRIDE * sizeof(float);  // 33,280 (aliased)

    dim3 grid(BS_TOTAL / MT);   // 4096 blocks of 32 tokens
    dim3 block(256);            // 8 waves

    const size_t packed_bytes = (size_t)(512 * 256 + 1280 * 256) * sizeof(bf16_t); // 917,504
    const bool packed = (d_ws != nullptr) && (ws_size >= packed_bytes);

    if (packed) {
        bf16_t* Pg = (bf16_t*)d_ws;
        bf16_t* Pc = Pg + 512 * 256;
        const int totG = 16 * KT_G * 32;   // 8192 frag-lanes
        const int totC = 16 * KT_C * 32;   // 20480 frag-lanes
        pack_weights_kernel<<<(totG + 255) / 256, 256, 0, stream>>>(graph_W, Pg, KT_G);
        pack_weights_kernel<<<(totC + 255) / 256, 256, 0, stream>>>(comb_W, Pc, KT_C);
        lgcn_fused_kernel<true><<<grid, block, lds_bytes, stream>>>(
            cate_testId, cate_itemId, cate_tag, cont_feats, interaction,
            emb_interaction, emb_testId, emb_itemId, emb_tag,
            graph_table, graph_W, graph_b,
            comb_W, comb_b, comb_g, comb_beta,
            cont_W, cont_b, cont_g, cont_beta, Pg, Pc, out);
    } else {
        lgcn_fused_kernel<false><<<grid, block, lds_bytes, stream>>>(
            cate_testId, cate_itemId, cate_tag, cont_feats, interaction,
            emb_interaction, emb_testId, emb_itemId, emb_tag,
            graph_table, graph_W, graph_b,
            comb_W, comb_b, comb_g, comb_beta,
            cont_W, cont_b, cont_g, cont_beta,
            (const bf16_t*)nullptr, (const bf16_t*)nullptr, out);
    }
}
